// FFWRelativeCrossAttentionModule_23922967839482
// MI455X (gfx1250) — compile-verified, matrix operand order checked by reading, not run on hardware
//
#include <hip/hip_runtime.h>

typedef __attribute__((ext_vector_type(16))) _Float16 v16h;
typedef __attribute__((ext_vector_type(8)))  float    v8f;

namespace {

constexpr int kB  = 4;
constexpr int kC  = 512;
constexpr int kH  = 8;
constexpr int kHD = 64;
constexpr int kL  = 4;

__device__ __forceinline__ v8f wmma_f16(v16h a, v16h b, v8f c) {
  // D = A(16x32 f16) * B(32x16 f16) + C(16x16 f32)
  return __builtin_amdgcn_wmma_f32_16x16x32_f16(
      /*neg_a=*/false, a, /*neg_b=*/false, b,
      /*c_mod=*/(short)0, c, /*reuse_a=*/false, /*reuse_b=*/false);
}

// ---------------- elementwise helpers ----------------

__global__ void f32_to_f16_kernel(const float* __restrict__ in,
                                  _Float16* __restrict__ out, size_t n) {
  size_t i = (size_t)blockIdx.x * blockDim.x + threadIdx.x;
  size_t stride = (size_t)gridDim.x * blockDim.x;
  for (; i < n; i += stride) out[i] = (_Float16)in[i];
}

// mod[b, o] = sum_c silu(t[b,c]) * w[o,c] + bias[o]   (w is (2C, C))
__global__ void mod_kernel(const float* __restrict__ t,
                           const float* __restrict__ w,
                           const float* __restrict__ bias,
                           float* __restrict__ mod) {
  int o = blockIdx.x * blockDim.x + threadIdx.x;
  if (o >= kB * 2 * kC) return;
  int b  = o / (2 * kC);
  int oc = o % (2 * kC);
  const float* tr = t + (size_t)b * kC;
  const float* wr = w + (size_t)oc * kC;
  float s = 0.f;
  for (int c = 0; c < kC; ++c) {
    float x = tr[c];
    s += (x / (1.f + expf(-x))) * wr[c];   // silu
  }
  mod[o] = s + bias[oc];
}

// x: (N,B,C) f32; out = x*(1+scale)+shift -> f16 (and optionally f32)
__global__ void adaln_kernel(const float* __restrict__ x,
                             const float* __restrict__ mod,
                             _Float16* __restrict__ outh,
                             float* __restrict__ outf, size_t n) {
  size_t i = (size_t)blockIdx.x * blockDim.x + threadIdx.x;
  size_t stride = (size_t)gridDim.x * blockDim.x;
  for (; i < n; i += stride) {
    int    c = (int)(i % kC);
    size_t m = i / kC;
    int    b = (int)(m % kB);
    float sc = mod[(size_t)b * 2 * kC + c];
    float sh = mod[(size_t)b * 2 * kC + kC + c];
    float v  = x[i] * (1.f + sc) + sh;
    outh[i] = (_Float16)v;
    if (outf) outf[i] = v;
  }
}

// x: (N,B,C) f32; pos: (B,N,C,2); out: (B,N,C) f16 rotary embedded
__global__ void rotary_kernel(const float* __restrict__ x,
                              const float* __restrict__ pos,
                              _Float16* __restrict__ out, int N, size_t n) {
  size_t i = (size_t)blockIdx.x * blockDim.x + threadIdx.x;
  size_t stride = (size_t)gridDim.x * blockDim.x;
  for (; i < n; i += stride) {
    int    c  = (int)(i % kC);
    size_t m  = i / kC;
    int    b  = (int)(m % kB);
    int    nn = (int)(m / kB);
    float xv = x[i];
    float x2 = (c & 1) ? x[i - 1] : -x[i + 1];
    size_t pidx = ((((size_t)b * N + nn) * kC) + c) * 2;
    float cs = pos[pidx], sn = pos[pidx + 1];
    out[((size_t)b * N + nn) * kC + c] = (_Float16)(xv * cs + x2 * sn);
  }
}

// vp: (Nkv,B,C) f32 -> vT: (B,H,HD,Nkv) f16
__global__ void vtrans_kernel(const float* __restrict__ x,
                              _Float16* __restrict__ out, int Nkv, size_t n) {
  size_t i = (size_t)blockIdx.x * blockDim.x + threadIdx.x;
  size_t stride = (size_t)gridDim.x * blockDim.x;
  for (; i < n; i += stride) {
    int    c  = (int)(i % kC);
    size_t m  = i / kC;
    int    b  = (int)(m % kB);
    int    kv = (int)(m / kB);
    int h = c / kHD, d = c % kHD;
    out[(((size_t)(b * kH + h)) * kHD + d) * Nkv + kv] = (_Float16)x[i];
  }
}

// one wave (32 lanes) per row of length C=512: out = LN(x+y)*g + b
__global__ void residual_ln_kernel(const float* __restrict__ x,
                                   const float* __restrict__ y,
                                   const float* __restrict__ g,
                                   const float* __restrict__ bt,
                                   float* __restrict__ outq,
                                   float* __restrict__ out2) {
  int row  = blockIdx.x;
  int lane = threadIdx.x & 31;
  const float* xr = x + (size_t)row * kC;
  const float* yr = y + (size_t)row * kC;
  float vals[16];
  float s = 0.f;
#pragma unroll
  for (int j = 0; j < 16; ++j) {
    int c = lane + 32 * j;
    float v = xr[c] + yr[c];
    vals[j] = v;
    s += v;
  }
#pragma unroll
  for (int m = 1; m < 32; m <<= 1) s += __shfl_xor(s, m, 32);
  float mean = s * (1.f / kC);
  float vs = 0.f;
#pragma unroll
  for (int j = 0; j < 16; ++j) {
    float d = vals[j] - mean;
    vs += d * d;
  }
#pragma unroll
  for (int m = 1; m < 32; m <<= 1) vs += __shfl_xor(vs, m, 32);
  float inv = rsqrtf(vs * (1.f / kC) + 1e-5f);
#pragma unroll
  for (int j = 0; j < 16; ++j) {
    int c = lane + 32 * j;
    float o = (vals[j] - mean) * inv * g[c] + bt[c];
    outq[(size_t)row * kC + c] = o;
    if (out2) out2[(size_t)row * kC + c] = o;
  }
}

// ---------------- WMMA GEMM: D = act((A @ W^T + bias) * scale) ----------------
// A: (M,K) f16 row-major; W: (N,K) f16 row-major.
// Register-blocked: one wave computes a 32x64 tile of D (2 M-subtiles x
// 4 N-subtiles -> 8 v_wmma per 32-wide K step on 6 fragment loads, ~21 FLOP/B).
__global__ void gemm_wmma_kernel(const _Float16* __restrict__ A,
                                 const _Float16* __restrict__ W,
                                 const float* __restrict__ bias,
                                 float* __restrict__ Df32,
                                 _Float16* __restrict__ Df16,
                                 int M, int N, int K, float scale, int relu) {
  int wave = (int)((blockIdx.x * blockDim.x + threadIdx.x) >> 5);
  int lane = threadIdx.x & 31;
  int ntiles = N >> 6;                 // 64-wide N blocks
  int tiles  = (M >> 5) * ntiles;      // 32-tall M blocks
  if (wave >= tiles) return;
  int tM = wave / ntiles, tN = wave % ntiles;
  int half = lane >> 4;
  int l16  = lane & 15;

  const _Float16* ap0 = A + (size_t)(tM * 32 + l16) * K + half * 16;
  const _Float16* ap1 = ap0 + (size_t)16 * K;
  const _Float16* bp0 = W + (size_t)(tN * 64 + l16) * K + half * 16;
  const size_t bstep = (size_t)16 * K;

  v8f acc[2][4];
#pragma unroll
  for (int mi = 0; mi < 2; ++mi)
#pragma unroll
    for (int j = 0; j < 4; ++j) acc[mi][j] = (v8f){};

  for (int kb = 0; kb < K; kb += 32) {
    __builtin_prefetch(ap0 + kb + 128, 0, 1);
    __builtin_prefetch(bp0 + kb + 128, 0, 1);
    v16h a0 = *reinterpret_cast<const v16h*>(ap0 + kb);
    v16h a1 = *reinterpret_cast<const v16h*>(ap1 + kb);
#pragma unroll
    for (int j = 0; j < 4; ++j) {
      v16h b = *reinterpret_cast<const v16h*>(bp0 + (size_t)j * bstep + kb);
      acc[0][j] = wmma_f16(a0, b, acc[0][j]);
      acc[1][j] = wmma_f16(a1, b, acc[1][j]);
    }
  }

#pragma unroll
  for (int j = 0; j < 4; ++j) {
    int n = tN * 64 + j * 16 + l16;
    float bv = bias ? bias[n] : 0.f;
#pragma unroll
    for (int mi = 0; mi < 2; ++mi) {
#pragma unroll
      for (int r = 0; r < 8; ++r) {
        int m = tM * 32 + mi * 16 + r + half * 8;
        float v = (acc[mi][j][r] + bv) * scale;
        if (relu) v = fmaxf(v, 0.f);
        if (Df32) Df32[(size_t)m * N + n] = v;
        if (Df16) Df16[(size_t)m * N + n] = (_Float16)v;
      }
    }
  }
}

// ---------------- flash attention, one wave per (b, h, 16-row q tile) ---------
// qf: (B,Nq,C) f16 (pre-scaled by HD^-0.5), kf: (B,Nkv,C) f16,
// vT: (B,H,HD,Nkv) f16, of: (Nq,B,C) f16
__global__ void attn_wmma_kernel(const _Float16* __restrict__ qf,
                                 const _Float16* __restrict__ kf,
                                 const _Float16* __restrict__ vT,
                                 _Float16* __restrict__ of,
                                 int Nq, int Nkv) {
  const int qt = blockIdx.x, h = blockIdx.y, b = blockIdx.z;
  const int lane = threadIdx.x & 31;
  const int half = lane >> 4, l16 = lane & 15;
  __shared__ _Float16 pbuf[16 * 32];

  const size_t qbase = (((size_t)b * Nq) + qt * 16 + l16) * kC + h * kHD + half * 16;
  v16h qa0 = *reinterpret_cast<const v16h*>(qf + qbase);
  v16h qa1 = *reinterpret_cast<const v16h*>(qf + qbase + 32);

  v8f o0 = {}, o1 = {}, o2 = {}, o3 = {};
  float rowm[8], rowl[8];
#pragma unroll
  for (int r = 0; r < 8; ++r) { rowm[r] = -1e30f; rowl[r] = 0.f; }

  const size_t vbase = ((size_t)(b * kH + h)) * kHD * (size_t)Nkv;

  for (int kv0 = 0; kv0 < Nkv; kv0 += 32) {
    // S = Q @ K^T for 16 q rows x 32 kv cols (two 16x16 tiles, K=64 in 2 steps)
    v8f s0 = {}, s1 = {};
    {
      const size_t k0 = (((size_t)b * Nkv) + kv0 + l16) * kC + h * kHD + half * 16;
      v16h kb0 = *reinterpret_cast<const v16h*>(kf + k0);
      v16h kb1 = *reinterpret_cast<const v16h*>(kf + k0 + 32);
      s0 = wmma_f16(qa0, kb0, s0);
      s0 = wmma_f16(qa1, kb1, s0);
      const size_t k1 = k0 + (size_t)16 * kC;
      v16h kc0 = *reinterpret_cast<const v16h*>(kf + k1);
      v16h kc1 = *reinterpret_cast<const v16h*>(kf + k1 + 32);
      s1 = wmma_f16(qa0, kc0, s1);
      s1 = wmma_f16(qa1, kc1, s1);
    }
    // online softmax (row stats reduced across the 16 lanes of each half)
    float fac[8];
#pragma unroll
    for (int r = 0; r < 8; ++r) {
      float mx = fmaxf(s0[r], s1[r]);
#pragma unroll
      for (int m = 1; m < 16; m <<= 1) mx = fmaxf(mx, __shfl_xor(mx, m, 32));
      float mnew = fmaxf(rowm[r], mx);
      float f    = expf(rowm[r] - mnew);
      float p0   = expf(s0[r] - mnew);
      float p1   = expf(s1[r] - mnew);
      float ps   = p0 + p1;
#pragma unroll
      for (int m = 1; m < 16; m <<= 1) ps += __shfl_xor(ps, m, 32);
      rowl[r] = rowl[r] * f + ps;
      rowm[r] = mnew;
      fac[r]  = f;
      pbuf[(r + 8 * half) * 32 + l16]      = (_Float16)p0;
      pbuf[(r + 8 * half) * 32 + 16 + l16] = (_Float16)p1;
    }
#pragma unroll
    for (int r = 0; r < 8; ++r) {
      o0[r] *= fac[r]; o1[r] *= fac[r]; o2[r] *= fac[r]; o3[r] *= fac[r];
    }
    __builtin_amdgcn_wave_barrier();
    // P in A-fragment layout via LDS bounce (C/D layout -> A layout transpose)
    v16h pa = *reinterpret_cast<const v16h*>(&pbuf[l16 * 32 + half * 16]);
    // O += P @ V for four 16-col d tiles (contiguous over kv thanks to vT)
    {
      v16h vb0 = *reinterpret_cast<const v16h*>(vT + vbase + (size_t)(0 + l16) * Nkv + kv0 + half * 16);
      o0 = wmma_f16(pa, vb0, o0);
      v16h vb1 = *reinterpret_cast<const v16h*>(vT + vbase + (size_t)(16 + l16) * Nkv + kv0 + half * 16);
      o1 = wmma_f16(pa, vb1, o1);
      v16h vb2 = *reinterpret_cast<const v16h*>(vT + vbase + (size_t)(32 + l16) * Nkv + kv0 + half * 16);
      o2 = wmma_f16(pa, vb2, o2);
      v16h vb3 = *reinterpret_cast<const v16h*>(vT + vbase + (size_t)(48 + l16) * Nkv + kv0 + half * 16);
      o3 = wmma_f16(pa, vb3, o3);
    }
    __builtin_amdgcn_wave_barrier();
  }
  // epilogue: normalize and write (Nq,B,C) layout at column h*HD
#pragma unroll
  for (int r = 0; r < 8; ++r) {
    float inv = 1.f / rowl[r];
    int row = qt * 16 + r + 8 * half;
    size_t obase = ((size_t)row * kB + b) * kC + h * kHD;
    of[obase + 0  + l16] = (_Float16)(o0[r] * inv);
    of[obase + 16 + l16] = (_Float16)(o1[r] * inv);
    of[obase + 32 + l16] = (_Float16)(o2[r] * inv);
    of[obase + 48 + l16] = (_Float16)(o3[r] * inv);
  }
}

}  // namespace

// ---------------------------------------------------------------------------

extern "C" void kernel_launch(void* const* d_in, const int* in_sizes, int n_in,
                              void* d_out, int out_size, void* d_ws, size_t ws_size,
                              hipStream_t stream) {
  (void)n_in; (void)out_size; (void)ws_size;
  const float* query     = (const float*)d_in[0];
  const float* value     = (const float*)d_in[1];
  const float* diff_ts   = (const float*)d_in[2];
  const float* query_pos = (const float*)d_in[3];
  const float* value_pos = (const float*)d_in[4];
  const float* aw_attn   = (const float*)d_in[5];
  const float* ab_attn   = (const float*)d_in[6];
  const float* in_w      = (const float*)d_in[7];
  const float* in_b      = (const float*)d_in[8];
  const float* out_w     = (const float*)d_in[9];
  const float* out_b     = (const float*)d_in[10];
  const float* ln1_g     = (const float*)d_in[11];
  const float* ln1_b     = (const float*)d_in[12];
  const float* aw_ffn    = (const float*)d_in[13];
  const float* ab_ffn    = (const float*)d_in[14];
  const float* w1        = (const float*)d_in[15];
  const float* b1        = (const float*)d_in[16];
  const float* w2        = (const float*)d_in[17];
  const float* b2        = (const float*)d_in[18];
  const float* ln2_g     = (const float*)d_in[19];
  const float* ln2_b     = (const float*)d_in[20];
  float* out = (float*)d_out;

  const int Nq  = in_sizes[0] / (kB * kC);   // 1024
  const int Nkv = in_sizes[1] / (kB * kC);   // 2048
  const int MQ  = Nq * kB;                   // 4096 rows
  const int MKV = Nkv * kB;                  // 8192 rows
  const float scaling = rsqrtf((float)kHD);  // HD^-0.5

  // ---- workspace bump allocator ----
  char* ws = (char*)d_ws;
  size_t off = 0;
  auto alloc = [&](size_t bytes) -> void* {
    void* p = ws + off;
    off += (bytes + 255) & ~(size_t)255;
    return p;
  };
  _Float16* w_in_h  = (_Float16*)alloc((size_t)kL * 3 * kC * kC * 2);
  _Float16* w_out_h = (_Float16*)alloc((size_t)kL * kC * kC * 2);
  _Float16* w1_h    = (_Float16*)alloc((size_t)kL * kC * kC * 2);
  _Float16* w2_h    = (_Float16*)alloc((size_t)kL * kC * kC * 2);
  _Float16* val_h   = (_Float16*)alloc((size_t)MKV * kC * 2);
  float*    q_cur   = (float*)   alloc((size_t)MQ * kC * 4);
  float*    mod_a   = (float*)   alloc((size_t)kB * 2 * kC * 4);
  float*    mod_f   = (float*)   alloc((size_t)kB * 2 * kC * 4);
  _Float16* act_h   = (_Float16*)alloc((size_t)MQ * kC * 2);   // aq / xa f16
  _Float16* h_h     = (_Float16*)alloc((size_t)MQ * kC * 2);   // attn-out / ffn hidden f16
  float*    proj    = (float*)   alloc((size_t)MKV * kC * 4);  // qp/kp/vp f32 scratch
  _Float16* qf16    = (_Float16*)alloc((size_t)kB * Nq * kC * 2);
  _Float16* kf16    = (_Float16*)alloc((size_t)kB * Nkv * kC * 2);
  _Float16* vT      = (_Float16*)alloc((size_t)kB * kC * Nkv * 2);
  float*    oacc    = (float*)   alloc((size_t)MQ * kC * 4);   // oproj / ffn out f32
  float*    xa_f    = (float*)   alloc((size_t)MQ * kC * 4);

  const int TB = 256;
  auto gsblocks = [](size_t n) { int b = (int)((n + 1023) / 1024); return b < 2048 ? b : 2048; };

  // ---- one-time (per call) conversions & state init ----
  hipMemcpyAsync(q_cur, query, (size_t)MQ * kC * 4, hipMemcpyDeviceToDevice, stream);
  {
    size_t n;
    n = (size_t)kL * 3 * kC * kC;
    f32_to_f16_kernel<<<gsblocks(n), TB, 0, stream>>>(in_w, w_in_h, n);
    n = (size_t)kL * kC * kC;
    f32_to_f16_kernel<<<gsblocks(n), TB, 0, stream>>>(out_w, w_out_h, n);
    f32_to_f16_kernel<<<gsblocks(n), TB, 0, stream>>>(w1, w1_h, n);
    f32_to_f16_kernel<<<gsblocks(n), TB, 0, stream>>>(w2, w2_h, n);
    n = (size_t)MKV * kC;
    f32_to_f16_kernel<<<gsblocks(n), TB, 0, stream>>>(value, val_h, n);
  }

  const size_t nQ  = (size_t)MQ * kC;
  const size_t nKV = (size_t)MKV * kC;
  // 32x64 output tile per wave, 8 waves (256 threads) per block
  const int gemmQ_blocks  = ((MQ / 32) * (kC / 64) + 7) / 8;
  const int gemmKV_blocks = ((MKV / 32) * (kC / 64) + 7) / 8;

  for (int i = 0; i < kL; ++i) {
    const _Float16* wq = w_in_h + (size_t)i * 3 * kC * kC;
    const _Float16* wk = wq + (size_t)kC * kC;
    const _Float16* wv = wk + (size_t)kC * kC;
    const float* bq = in_b + (size_t)i * 3 * kC;
    const float* bk = bq + kC;
    const float* bv = bk + kC;

    // AdaLN modulations for this layer
    mod_kernel<<<(kB * 2 * kC + TB - 1) / TB, TB, 0, stream>>>(
        diff_ts, aw_attn + (size_t)i * 2 * kC * kC, ab_attn + (size_t)i * 2 * kC, mod_a);
    mod_kernel<<<(kB * 2 * kC + TB - 1) / TB, TB, 0, stream>>>(
        diff_ts, aw_ffn + (size_t)i * 2 * kC * kC, ab_ffn + (size_t)i * 2 * kC, mod_f);

    // aq = adaln(q) -> f16
    adaln_kernel<<<gsblocks(nQ), TB, 0, stream>>>(q_cur, mod_a, act_h, (float*)nullptr, nQ);

    // qp = (aq @ wq^T + bq) * scaling ; rotary -> qf16 (B,Nq,C)
    gemm_wmma_kernel<<<gemmQ_blocks, TB, 0, stream>>>(
        act_h, wq, bq, proj, (_Float16*)nullptr, MQ, kC, kC, scaling, 0);
    rotary_kernel<<<gsblocks(nQ), TB, 0, stream>>>(proj, query_pos, qf16, Nq, nQ);

    // kp = value @ wk^T + bk ; rotary -> kf16 (B,Nkv,C)
    gemm_wmma_kernel<<<gemmKV_blocks, TB, 0, stream>>>(
        val_h, wk, bk, proj, (_Float16*)nullptr, MKV, kC, kC, 1.f, 0);
    rotary_kernel<<<gsblocks(nKV), TB, 0, stream>>>(proj, value_pos, kf16, Nkv, nKV);

    // vp = value @ wv^T + bv ; transpose -> vT (B,H,HD,Nkv)
    gemm_wmma_kernel<<<gemmKV_blocks, TB, 0, stream>>>(
        val_h, wv, bv, proj, (_Float16*)nullptr, MKV, kC, kC, 1.f, 0);
    vtrans_kernel<<<gsblocks(nKV), TB, 0, stream>>>(proj, vT, Nkv, nKV);

    // attention -> h_h (Nq,B,C) f16
    attn_wmma_kernel<<<dim3(Nq / 16, kH, kB), 32, 0, stream>>>(qf16, kf16, vT, h_h, Nq, Nkv);

    // out proj -> oacc f32 ; q = LN(q + o)
    gemm_wmma_kernel<<<gemmQ_blocks, TB, 0, stream>>>(
        h_h, w_out_h + (size_t)i * kC * kC, out_b + (size_t)i * kC,
        oacc, (_Float16*)nullptr, MQ, kC, kC, 1.f, 0);
    residual_ln_kernel<<<MQ, 32, 0, stream>>>(
        q_cur, oacc, ln1_g + (size_t)i * kC, ln1_b + (size_t)i * kC, q_cur, (float*)nullptr);

    // FFN: xa = adaln(q) (keep f32 for residual), h = relu(xa@w1^T+b1) f16
    adaln_kernel<<<gsblocks(nQ), TB, 0, stream>>>(q_cur, mod_f, act_h, xa_f, nQ);
    gemm_wmma_kernel<<<gemmQ_blocks, TB, 0, stream>>>(
        act_h, w1_h + (size_t)i * kC * kC, b1 + (size_t)i * kC,
        (float*)nullptr, h_h, MQ, kC, kC, 1.f, 1);
    gemm_wmma_kernel<<<gemmQ_blocks, TB, 0, stream>>>(
        h_h, w2_h + (size_t)i * kC * kC, b2 + (size_t)i * kC,
        oacc, (_Float16*)nullptr, MQ, kC, kC, 1.f, 0);

    // q = LN(xa + h) ; also write layer output slice
    residual_ln_kernel<<<MQ, 32, 0, stream>>>(
        xa_f, oacc, ln2_g + (size_t)i * kC, ln2_b + (size_t)i * kC,
        q_cur, out + (size_t)i * MQ * kC);
  }
}
